// GFLayerNSF_76656576299114
// MI455X (gfx1250) — matrix operand order repaired and checked by reading, not compiled, not run on hardware
//
#include <hip/hip_runtime.h>
#include <hip/hip_bf16.h>

// ---------------------------------------------------------------------------
// MI455X (gfx1250) GF-layer NSF forward pass.
// GEMMs (Wconv, MLP, QKV, Wo) and both attention einsums -> v_wmma_f32_16x16x32_bf16.
// Flash attention (online softmax).  All WMMA fragments are contiguous 16B
// loads (ds_load_b128 / global b128).  K tiles in attention are staged into
// LDS by the Tensor Data Mover (tensor_load_to_lds + s_wait_tensorcnt) when
// the toolchain exposes the builtin; otherwise direct global fragment loads.
// ---------------------------------------------------------------------------

typedef __attribute__((ext_vector_type(16))) __bf16          bf16x16;
typedef __attribute__((ext_vector_type(8)))  float           f32x8;
typedef __attribute__((ext_vector_type(4)))  float           f32x4;
typedef __attribute__((ext_vector_type(16))) unsigned short  u16x16;
typedef __attribute__((ext_vector_type(8)))  unsigned short  u16x8;
typedef __attribute__((ext_vector_type(4)))  unsigned short  u16x4;
typedef __attribute__((ext_vector_type(4)))  unsigned int    u32x4;
typedef __attribute__((ext_vector_type(8)))  int             i32x8;
typedef __attribute__((ext_vector_type(4)))  int             i32x4;

#define E_DIM   64
#define HALF    32
#define HID     64
#define VDIM    448      // (3*K_BINS-1)*half
#define SEQ_N   2048
#define BATCH   8
#define MROWS   (BATCH * SEQ_N)   // 16384
#define KBINS   5
#define TAILB   3.0f
#define MIN_BW  0.001f
#define MIN_BH  0.001f
#define MIN_D   0.001f

#if __has_builtin(__builtin_amdgcn_tensor_load_to_lds)
#define HAVE_TDM 1
#else
#define HAVE_TDM 0
#endif

__device__ __forceinline__ unsigned short f2bf(float f) {
  unsigned u = __builtin_bit_cast(unsigned, f);
  unsigned r = (u + 0x7FFFu + ((u >> 16) & 1u)) >> 16;   // RNE
  return (unsigned short)r;
}

__device__ __forceinline__ float softplusf(float x) {
  return fmaxf(x, 0.f) + log1pf(__expf(-fabsf(x)));
}

__device__ __forceinline__ bf16x16 mk_frag(u16x8 lo, u16x8 hi) {
  u16x16 c = __builtin_shufflevector(lo, hi, 0, 1, 2, 3, 4, 5, 6, 7,
                                             8, 9, 10, 11, 12, 13, 14, 15);
  return __builtin_bit_cast(bf16x16, c);
}

// A fragment (16x32 bf16, MxK) from row-major source (stride in elems).
__device__ __forceinline__ bf16x16 fragA(const unsigned short* p, int stride) {
  int lane = threadIdx.x & 31;
  int m  = lane & 15;
  int kb = (lane >> 4) << 3;
  u16x8 lo = *(const u16x8*)&p[m * stride + kb];
  u16x8 hi = *(const u16x8*)&p[m * stride + 16 + kb];
  return mk_frag(lo, hi);
}

// B fragment (32x16 bf16, KxN) from a TRANSPOSED [n][k] tile (stride 32).
__device__ __forceinline__ bf16x16 fragBT(const unsigned short* p, int ncol) {
  int lane = threadIdx.x & 31;
  int n  = (lane & 15) + ncol;
  int kb = (lane >> 4) << 3;
  u16x8 lo = *(const u16x8*)&p[n * 32 + kb];
  u16x8 hi = *(const u16x8*)&p[n * 32 + 16 + kb];
  return mk_frag(lo, hi);
}

__device__ __forceinline__ void wait_tensorcnt0() {
#if __has_builtin(__builtin_amdgcn_s_wait_tensorcnt)
  __builtin_amdgcn_s_wait_tensorcnt(0);
#else
  asm volatile("s_wait_tensorcnt 0x0" ::: "memory");
#endif
}

#if HAVE_TDM
// Issue a 2D TDM load: tile_d0 x tile_d1 elements of 2 bytes, row stride
// stride_d0 (elements), from gaddr into LDS byte offset lds_off.
// D# layout per CDNA5 ISA section 8 (group0 128b, group1 256b, groups 2/3 zero).
__device__ __forceinline__ void tdm_load_2d(
    unsigned lds_off, const void* gaddr,
    unsigned tile_d0, unsigned tile_d1,
    unsigned tens_d0, unsigned tens_d1, unsigned stride_d0)
{
  unsigned long long ga = (unsigned long long)(size_t)gaddr;
  u32x4 g0;
  g0[0] = 1u;                                              // count=1
  g0[1] = lds_off;                                         // lds_addr
  g0[2] = (unsigned)(ga & 0xFFFFFFFFu);                    // global_addr lo
  g0[3] = (unsigned)((ga >> 32) & 0x1FFFFFFu) | (2u << 30); // addr hi | type=2
  i32x8 g1;
  g1[0] = (int)(1u << 16);                                 // data_size=1 (2B)
  g1[1] = (int)(tens_d0 << 16);                            // tensor_dim0[15:0]
  g1[2] = (int)((tens_d0 >> 16) | (tens_d1 << 16));        // dim0 hi | dim1 lo
  g1[3] = (int)((tens_d1 >> 16) | (tile_d0 << 16));        // dim1 hi | tile_dim0
  g1[4] = (int)tile_d1;                                    // tile_dim1
  g1[5] = (int)stride_d0;                                  // dim0_stride lo
  g1[6] = 0;
  g1[7] = 0;
  i32x4 z4 = {0, 0, 0, 0};
#if defined(__clang_major__) && __clang_major__ >= 23
  i32x8 z8 = {0, 0, 0, 0, 0, 0, 0, 0};
  __builtin_amdgcn_tensor_load_to_lds(g0, g1, z4, z4, z8, 0);
#else
  __builtin_amdgcn_tensor_load_to_lds(g0, g1, z4, z4, 0);
#endif
}
#endif

// ---------------------------------------------------------------------------
// Generic WMMA GEMM: C = [resid +] act(A(f32,MxK) @ W(f32,KxN) + bias)
// Block: 256 thr (8 waves), tile 128(M) x 64(N), K staged in 32-chunks.
// Optional outputs: f32, bf16, and per-batch-transposed bf16 (attention V).
// ---------------------------------------------------------------------------
__global__ __launch_bounds__(256) void k_gemm(
    const float* __restrict__ A, int lda,
    const float* __restrict__ W, int ldw,
    const float* __restrict__ bias,
    const float* __restrict__ resid,
    float* __restrict__ Cf, unsigned short* __restrict__ Cbf,
    unsigned short* __restrict__ CbfT, int ldc,
    int K, int relu)
{
  __shared__ __align__(16) unsigned short sA [128 * 32];
  __shared__ __align__(16) unsigned short sBT[64 * 32];    // transposed [n][k]
  const int t  = threadIdx.x;
  const int w  = t >> 5;
  const int m0 = blockIdx.x * 128;
  const int n0 = blockIdx.y * 64;

  f32x8 acc[4];
#pragma unroll
  for (int i = 0; i < 4; ++i) {
#pragma unroll
    for (int r = 0; r < 8; ++r) acc[i][r] = 0.f;
  }

  for (int k0 = 0; k0 < K; k0 += 32) {
    // A tile 128x32: vector f32x4 reads, u16x4 LDS stores
#pragma unroll
    for (int i = 0; i < 4; ++i) {                   // 1024 quads / 256 thr
      int q = t + i * 256;
      int r = q >> 3, c4 = (q & 7) * 4;
      f32x4 v = *(const f32x4*)&A[(size_t)(m0 + r) * lda + k0 + c4];
      u16x4 h;
#pragma unroll
      for (int j = 0; j < 4; ++j) h[j] = f2bf(v[j]);
      *(u16x4*)&sA[r * 32 + c4] = h;
    }
    // W tile 32x64 -> transposed LDS [n][k]
#pragma unroll
    for (int i = 0; i < 2; ++i) {                   // 512 quads / 256 thr
      int q = t + i * 256;
      int r = q >> 4, c4 = (q & 15) * 4;
      f32x4 v = *(const f32x4*)&W[(size_t)(k0 + r) * ldw + n0 + c4];
#pragma unroll
      for (int j = 0; j < 4; ++j) sBT[(c4 + j) * 32 + r] = f2bf(v[j]);
    }
    __syncthreads();
    bf16x16 a = fragA(sA + (w * 16) * 32, 32);
#pragma unroll
    for (int nt = 0; nt < 4; ++nt) {
      bf16x16 b = fragBT(sBT, nt * 16);
      acc[nt] = __builtin_amdgcn_wmma_f32_16x16x32_bf16(
          false, a, false, b, (short)0, acc[nt], false, false);
    }
    __syncthreads();
  }

  const int lane = t & 31;
  const int nlo  = lane & 15;
  const int mhi  = (lane >> 4) * 8;
#pragma unroll
  for (int nt = 0; nt < 4; ++nt) {
    int n = n0 + nt * 16 + nlo;
    float bv = bias ? bias[n] : 0.f;
#pragma unroll
    for (int r = 0; r < 8; ++r) {
      int m = m0 + w * 16 + mhi + r;
      float v = acc[nt][r] + bv;
      if (relu)  v = fmaxf(v, 0.f);
      if (resid) v += resid[(size_t)m * ldc + n];
      if (Cf)   Cf  [(size_t)m * ldc + n] = v;
      if (Cbf)  Cbf [(size_t)m * ldc + n] = f2bf(v);
      if (CbfT) CbfT[((size_t)(m >> 11) * VDIM + n) * SEQ_N + (m & (SEQ_N - 1))] = f2bf(v);
    }
  }
}

// ---------------------------------------------------------------------------
// Flash attention + residual: O = Q + softmax(Q K^T / sqrt(448)) V
// Block: 128 thr (4 waves) per 16-query tile; 64-key tiles.
// Q fragments resident in registers; K tiles TDM-staged into per-wave LDS
// slices (fallback: direct global b128); V fragments direct global b128 from
// pre-transposed Vt[b][d][key].
// ---------------------------------------------------------------------------
__global__ __launch_bounds__(128) void k_attn(
    const unsigned short* __restrict__ Qbf,
    const unsigned short* __restrict__ Kbf,
    const unsigned short* __restrict__ Vt,   // [b][d][key]
    const float*          __restrict__ Qf,
    float*                __restrict__ O)
{
#if HAVE_TDM
  __shared__ __align__(16) unsigned short sK[4 * 16 * VDIM];  // 56 KB, per-wave
#endif
  __shared__ __align__(16) float          sS[16 * 64];
  __shared__ __align__(16) unsigned short sP[16 * 64];
  __shared__ float s_m[16], s_l[16], s_al[16];

  const int t    = threadIdx.x;
  const int w    = t >> 5;
  const int lane = t & 31;
  const int b    = blockIdx.x >> 7;
  const int qt   = blockIdx.x & 127;
  const int row0 = b * SEQ_N + qt * 16;
  const float scale = rsqrtf((float)VDIM);

  const int nlo = lane & 15;
  const int mhi = (lane >> 4) * 8;
  const int kbA = (lane >> 4) << 3;   // A-fragment K base (0/8)
  const int kbB = (lane >> 4) << 3;   // B-fragment K base (0/8)

  // Q fragments (14 x 8 VGPRs), identical in all 4 waves.
  bf16x16 qfrag[14];
#pragma unroll
  for (int kc = 0; kc < 14; ++kc) {
    const unsigned short* qp = &Qbf[(size_t)(row0 + nlo) * VDIM + kc * 32];
    qfrag[kc] = mk_frag(*(const u16x8*)&qp[kbA], *(const u16x8*)&qp[16 + kbA]);
  }

  f32x8 oacc[7];
#pragma unroll
  for (int i = 0; i < 7; ++i) {
#pragma unroll
    for (int r = 0; r < 8; ++r) oacc[i][r] = 0.f;
  }
  if (t < 16) { s_m[t] = -1.0e30f; s_l[t] = 0.f; }
  __syncthreads();

  for (int kt = 0; kt < SEQ_N; kt += 64) {
    if (kt + 64 < SEQ_N)
      __builtin_prefetch(&Kbf[(size_t)(b * SEQ_N + kt + 64) * VDIM], 0, 1);

#if HAVE_TDM
    // TDM: each wave DMAs its private 16-key x 448-d bf16 slice into LDS.
    // Prior ds reads of sK must have drained before the TDM overwrites it.
    asm volatile("s_wait_dscnt 0x0" ::: "memory");
    tdm_load_2d((unsigned)(size_t)&sK[(size_t)w * 16 * VDIM],
                &Kbf[(size_t)(b * SEQ_N + kt + w * 16) * VDIM],
                /*tile*/ VDIM, 16, /*tensor*/ VDIM, 16, /*stride*/ VDIM);
    wait_tensorcnt0();
    const unsigned short* krow = &sK[(size_t)(w * 16 + nlo) * VDIM];
#else
    const unsigned short* krow = &Kbf[(size_t)(b * SEQ_N + kt + w * 16 + nlo) * VDIM];
#endif

    // S tile: this wave's 16 keys.  B-frag = two contiguous b128 per lane.
    f32x8 s;
#pragma unroll
    for (int r = 0; r < 8; ++r) s[r] = 0.f;
#pragma unroll
    for (int kc = 0; kc < 14; ++kc) {
      const unsigned short* kp = &krow[kc * 32 + kbB];
      bf16x16 bf = mk_frag(*(const u16x8*)&kp[0], *(const u16x8*)&kp[16]);
      s = __builtin_amdgcn_wmma_f32_16x16x32_bf16(
          false, qfrag[kc], false, bf, (short)0, s, false, false);
    }
#pragma unroll
    for (int r = 0; r < 8; ++r)
      sS[(mhi + r) * 64 + w * 16 + nlo] = s[r] * scale;
    __syncthreads();

    // online softmax, one thread per query row
    if (t < 16) {
      float mOld = s_m[t], mNew = mOld;
      for (int j = 0; j < 64; ++j) mNew = fmaxf(mNew, sS[t * 64 + j]);
      float al = __expf(mOld - mNew), ssum = 0.f;
      for (int j = 0; j < 64; ++j) {
        float e = __expf(sS[t * 64 + j] - mNew);
        sP[t * 64 + j] = f2bf(e);
        ssum += e;
      }
      s_m[t] = mNew; s_l[t] = s_l[t] * al + ssum; s_al[t] = al;
    }
    __syncthreads();

    // rescale + accumulate O += P(16x64) @ V(64 x 112-slice)
    {
      float alr[8];
#pragma unroll
      for (int r = 0; r < 8; ++r) alr[r] = s_al[mhi + r];
#pragma unroll
      for (int i = 0; i < 7; ++i) {
#pragma unroll
        for (int r = 0; r < 8; ++r) oacc[i][r] *= alr[r];
      }
      const int d0 = w * 112;
#pragma unroll
      for (int kc = 0; kc < 2; ++kc) {
        const unsigned short* pp = &sP[nlo * 64 + kc * 32];
        bf16x16 pf = mk_frag(*(const u16x8*)&pp[kbA], *(const u16x8*)&pp[16 + kbA]);
#pragma unroll
        for (int nt = 0; nt < 7; ++nt) {
          int dcol = d0 + nt * 16 + nlo;
          const unsigned short* vp =
              &Vt[((size_t)(b * VDIM + dcol)) * SEQ_N + kt + kc * 32 + kbB];
          bf16x16 vf = mk_frag(*(const u16x8*)&vp[0], *(const u16x8*)&vp[16]);
          oacc[nt] = __builtin_amdgcn_wmma_f32_16x16x32_bf16(
              false, pf, false, vf, (short)0, oacc[nt], false, false);
        }
      }
    }
    __syncthreads();
  }

  // O = Q + acc / l
  {
    const int d0 = w * 112;
    float linv[8];
#pragma unroll
    for (int r = 0; r < 8; ++r) linv[r] = 1.f / s_l[mhi + r];
#pragma unroll
    for (int nt = 0; nt < 7; ++nt) {
      int dcol = d0 + nt * 16 + nlo;
#pragma unroll
      for (int r = 0; r < 8; ++r) {
        int row = row0 + mhi + r;
        O[(size_t)row * VDIM + dcol] =
            oacc[nt][r] * linv[r] + Qf[(size_t)row * VDIM + dcol];
      }
    }
  }
}

// ---------------------------------------------------------------------------
// RQS spline per element + per-batch logdet reduction
// ---------------------------------------------------------------------------
__global__ __launch_bounds__(256) void k_spline(
    const float* __restrict__ sab,            // 16384 x 448 (32 dims x 14)
    const float* __restrict__ tgt, int ldt,   // strided target
    float* __restrict__ y,                    // 16384 x 32
    float* __restrict__ ldsum)                // [8], atomicAdd
{
  __shared__ float red[256];
  int gid = blockIdx.x * 256 + threadIdx.x;
  int row = gid >> 5;
  int j   = gid & 31;
  int b   = row >> 11;
  const float* raw = sab + (size_t)row * VDIM + j * 14;

  float uw[KBINS], uh[KBINS], d4[4];
  // _coupling: W = 2T*softmax(raw[0:5]), H = 2T*softmax(raw[5:10]), D = softplus
  {
    float mw = -1e30f, mh = -1e30f;
    for (int c = 0; c < KBINS; ++c) { uw[c] = raw[c]; uh[c] = raw[KBINS + c];
                                      mw = fmaxf(mw, uw[c]); mh = fmaxf(mh, uh[c]); }
    float sw = 0.f, sh = 0.f;
    for (int c = 0; c < KBINS; ++c) { uw[c] = __expf(uw[c] - mw); sw += uw[c];
                                      uh[c] = __expf(uh[c] - mh); sh += uh[c]; }
    for (int c = 0; c < KBINS; ++c) { uw[c] = 2.f * TAILB * uw[c] / sw;
                                      uh[c] = 2.f * TAILB * uh[c] / sh; }
    for (int c = 0; c < 4; ++c) d4[c] = softplusf(raw[2 * KBINS + c]);
  }
  // _unconstrained_rqs
  const float cpad = logf(__expf(1.0f - MIN_D) - 1.0f);
  float derivs[KBINS + 1];
  derivs[0]     = MIN_D + softplusf(cpad);
  derivs[KBINS] = MIN_D + softplusf(cpad);
  for (int c = 1; c < KBINS; ++c) derivs[c] = MIN_D + softplusf(d4[c - 1]);

  float widths[KBINS], heights[KBINS], cumw[KBINS + 1], cumh[KBINS + 1];
  {
    float mw = -1e30f, mh = -1e30f;
    for (int c = 0; c < KBINS; ++c) { mw = fmaxf(mw, uw[c]); mh = fmaxf(mh, uh[c]); }
    float sw = 0.f, sh = 0.f;
    for (int c = 0; c < KBINS; ++c) { widths[c]  = __expf(uw[c] - mw); sw += widths[c];
                                      heights[c] = __expf(uh[c] - mh); sh += heights[c]; }
    float accw = 0.f, acch = 0.f;
    cumw[0] = -TAILB; cumh[0] = -TAILB;
    for (int c = 0; c < KBINS; ++c) {
      widths[c]  = MIN_BW + (1.f - MIN_BW * KBINS) * widths[c] / sw;
      heights[c] = MIN_BH + (1.f - MIN_BH * KBINS) * heights[c] / sh;
      accw += widths[c]; acch += heights[c];
      cumw[c + 1] = 2.f * TAILB * accw - TAILB;
      cumh[c + 1] = 2.f * TAILB * acch - TAILB;
    }
    cumw[KBINS] = TAILB; cumh[KBINS] = TAILB;
    for (int c = 0; c < KBINS; ++c) { widths[c]  = cumw[c + 1] - cumw[c];
                                      heights[c] = cumh[c + 1] - cumh[c]; }
  }

  float x  = tgt[(size_t)row * ldt + j];
  bool inside = (x >= -TAILB) && (x <= TAILB);
  float xc = fminf(fmaxf(x, -TAILB), TAILB);
  int cnt = 0;
  for (int c = 0; c <= KBINS; ++c) {
    float bin = (c == KBINS) ? (cumw[KBINS] + 1e-6f) : cumw[c];
    cnt += (xc >= bin) ? 1 : 0;
  }
  int idx = min(max(cnt - 1, 0), KBINS - 1);

  float in_cw = cumw[idx], in_w = widths[idx];
  float in_ch = cumh[idx], in_h = heights[idx];
  float delta = in_h / in_w;
  float dk = derivs[idx], dk1 = derivs[idx + 1];
  float th  = (xc - in_cw) / in_w;
  float t1m = th * (1.f - th);
  float numer = in_h * (delta * th * th + dk * t1m);
  float denom = delta + (dk + dk1 - 2.f * delta) * t1m;
  float yv = in_ch + numer / denom;
  float dnum = delta * delta * (dk1 * th * th + 2.f * delta * t1m + dk * (1.f - th) * (1.f - th));
  float ld = logf(dnum) - 2.f * logf(denom);
  yv = inside ? yv : x;
  ld = inside ? ld : 0.f;

  y[(size_t)row * 32 + j] = yv;

  red[threadIdx.x] = ld;
  __syncthreads();
  for (int s = 128; s > 0; s >>= 1) {
    if (threadIdx.x < s) red[threadIdx.x] += red[threadIdx.x + s];
    __syncthreads();
  }
  if (threadIdx.x == 0) atomicAdd(&ldsum[b], red[0]);
}

// ---------------------------------------------------------------------------
// Small utility kernels
// ---------------------------------------------------------------------------
__global__ void k_preaffine(const float* __restrict__ x, const float* __restrict__ mu,
                            const float* __restrict__ ls, float* __restrict__ z, int total) {
  int i = blockIdx.x * 256 + threadIdx.x;
  if (i < total) { int c = i & 63; z[i] = x[i] * __expf(ls[c]) + mu[c]; }
}

__global__ void k_slogdet(const float* __restrict__ W, float* __restrict__ out) {
  __shared__ float a[64 * 64];
  __shared__ float s_acc;
  __shared__ int   piv;
  int t = threadIdx.x;
  for (int i = t; i < 4096; i += blockDim.x) a[i] = W[i];
  if (t == 0) s_acc = 0.f;
  __syncthreads();
  for (int k = 0; k < 64; ++k) {
    if (t == 0) {
      int p = k; float best = fabsf(a[k * 64 + k]);
      for (int i = k + 1; i < 64; ++i) {
        float v = fabsf(a[i * 64 + k]);
        if (v > best) { best = v; p = i; }
      }
      piv = p;
      s_acc += logf(best + 1e-30f);
    }
    __syncthreads();
    if (piv != k && t < 64) {
      float tmp = a[k * 64 + t]; a[k * 64 + t] = a[piv * 64 + t]; a[piv * 64 + t] = tmp;
    }
    __syncthreads();
    for (int i = k + 1 + t; i < 64; i += blockDim.x) {
      float f = a[i * 64 + k] / a[k * 64 + k];
      for (int jj = k; jj < 64; ++jj) a[i * 64 + jj] -= f * a[k * 64 + jj];
    }
    __syncthreads();
  }
  if (t == 0) out[0] = s_acc;
}

__global__ void k_ldinit(const float* __restrict__ ls, const float* __restrict__ slog,
                         float* __restrict__ ld) {
  if (threadIdx.x < BATCH) {
    float s = 0.f;
    for (int c = 0; c < 64; ++c) s += ls[c];
    ld[threadIdx.x] = (s + slog[0]) * (float)SEQ_N;
  }
}

__global__ void k_pack(const float* __restrict__ lo, const float* __restrict__ up,
                       float* __restrict__ out, int rows) {
  int i = blockIdx.x * 256 + threadIdx.x;
  if (i < rows * 64) {
    int r = i >> 6, c = i & 63;
    out[i] = (c < 32) ? lo[r * 32 + c] : up[r * 32 + (c - 32)];
  }
}

// ---------------------------------------------------------------------------
// Host orchestration
// ---------------------------------------------------------------------------
struct CoupW {
  const float *W1, *b1, *W2, *b2, *W3, *b3;
  const float *Wq, *bq, *Wk, *bk, *Wv, *bv, *Wo, *bo;
};

struct Scratch {
  float *zpre, *z, *h1, *h2, *mo, *Qf, *Obuf, *sab, *up1, *lo1, *slog;
  unsigned short *Qbf, *Kbf, *Vt;
};

static void run_coupling(const CoupW& p, const float* cond, int lda,
                         const float* tgt, int ldt, float* yout,
                         float* ldout, const Scratch& s, hipStream_t st) {
  dim3 g1(MROWS / 128, 1), g7(MROWS / 128, 7), blk(256);
  // MLP
  k_gemm<<<g1, blk, 0, st>>>(cond, lda,  p.W1, HID, p.b1, nullptr, s.h1, nullptr, nullptr, HID, HALF, 1);
  k_gemm<<<g1, blk, 0, st>>>(s.h1, HID,  p.W2, HID, p.b2, nullptr, s.h2, nullptr, nullptr, HID, HID, 1);
  k_gemm<<<g1, blk, 0, st>>>(s.h2, HID,  p.W3, HID, p.b3, nullptr, s.mo, nullptr, nullptr, HID, HID, 0);
  // QKV projections (f32 Q for residual, bf16 copies for WMMA; V transposed)
  k_gemm<<<g7, blk, 0, st>>>(s.mo, HID, p.Wq, VDIM, p.bq, nullptr, s.Qf,   s.Qbf,  nullptr, VDIM, HID, 0);
  k_gemm<<<g7, blk, 0, st>>>(s.mo, HID, p.Wk, VDIM, p.bk, nullptr, nullptr, s.Kbf, nullptr, VDIM, HID, 0);
  k_gemm<<<g7, blk, 0, st>>>(s.mo, HID, p.Wv, VDIM, p.bv, nullptr, nullptr, nullptr, s.Vt, VDIM, HID, 0);
  // flash attention + residual
  k_attn<<<dim3(BATCH * SEQ_N / 16), dim3(128), 0, st>>>(s.Qbf, s.Kbf, s.Vt, s.Qf, s.Obuf);
  // out = O + relu(O @ Wo + bo)
  k_gemm<<<g7, blk, 0, st>>>(s.Obuf, VDIM, p.Wo, VDIM, p.bo, s.Obuf, s.sab, nullptr, nullptr, VDIM, VDIM, 1);
  // spline transform of tgt + logdet accumulation
  k_spline<<<dim3(MROWS * 32 / 256), blk, 0, st>>>(s.sab, tgt, ldt, yout, ldout);
}

extern "C" void kernel_launch(void* const* d_in, const int* in_sizes, int n_in,
                              void* d_out, int out_size, void* d_ws, size_t ws_size,
                              hipStream_t stream) {
  (void)in_sizes; (void)n_in; (void)out_size; (void)ws_size;
  const float* x     = (const float*)d_in[0];
  const float* mu    = (const float*)d_in[1];
  const float* lsig  = (const float*)d_in[2];
  const float* Wconv = (const float*)d_in[3];
  CoupW f1 = { (const float*)d_in[4],  (const float*)d_in[5],  (const float*)d_in[6],
               (const float*)d_in[7],  (const float*)d_in[8],  (const float*)d_in[9],
               (const float*)d_in[10], (const float*)d_in[11], (const float*)d_in[12],
               (const float*)d_in[13], (const float*)d_in[14], (const float*)d_in[15],
               (const float*)d_in[16], (const float*)d_in[17] };
  CoupW f2 = { (const float*)d_in[18], (const float*)d_in[19], (const float*)d_in[20],
               (const float*)d_in[21], (const float*)d_in[22], (const float*)d_in[23],
               (const float*)d_in[24], (const float*)d_in[25], (const float*)d_in[26],
               (const float*)d_in[27], (const float*)d_in[28], (const float*)d_in[29],
               (const float*)d_in[30], (const float*)d_in[31] };

  float* out_z  = (float*)d_out;                          // (8,2048,64)
  float* out_ld = (float*)d_out + (size_t)MROWS * E_DIM;  // (8,)

  // carve workspace (256B aligned)
  char* p = (char*)d_ws;
  auto carve = [&](size_t bytes) -> void* {
    void* r = (void*)p; p += (bytes + 255) & ~(size_t)255; return r;
  };
  Scratch s;
  s.zpre = (float*)carve((size_t)MROWS * E_DIM * 4);
  s.z    = (float*)carve((size_t)MROWS * E_DIM * 4);
  s.h1   = (float*)carve((size_t)MROWS * HID * 4);
  s.h2   = (float*)carve((size_t)MROWS * HID * 4);
  s.mo   = (float*)carve((size_t)MROWS * HID * 4);
  s.Qf   = (float*)carve((size_t)MROWS * VDIM * 4);
  s.Obuf = (float*)carve((size_t)MROWS * VDIM * 4);
  s.sab  = (float*)carve((size_t)MROWS * VDIM * 4);
  s.up1  = (float*)carve((size_t)MROWS * HALF * 4);
  s.lo1  = (float*)carve((size_t)MROWS * HALF * 4);
  s.Qbf  = (unsigned short*)carve((size_t)MROWS * VDIM * 2);
  s.Kbf  = (unsigned short*)carve((size_t)MROWS * VDIM * 2);
  s.Vt   = (unsigned short*)carve((size_t)MROWS * VDIM * 2);
  s.slog = (float*)carve(256);

  // log-det base: n*sum(log_sigma) + n*slogdet(Wconv)
  k_slogdet<<<1, 64, 0, stream>>>(Wconv, s.slog);
  k_ldinit<<<1, 32, 0, stream>>>(lsig, s.slog, out_ld);

  // z = (x*exp(log_sigma)+mu) @ Wconv
  k_preaffine<<<dim3(MROWS * E_DIM / 256), dim3(256), 0, stream>>>(
      x, mu, lsig, s.zpre, MROWS * E_DIM);
  k_gemm<<<dim3(MROWS / 128, 1), dim3(256), 0, stream>>>(
      s.zpre, E_DIM, Wconv, E_DIM, nullptr, nullptr, s.z, nullptr, nullptr, E_DIM, E_DIM, 0);

  // coupling 1: cond = lower = z[:, :32], tgt = upper = z[:, 32:]  -> up1
  run_coupling(f1, s.z, E_DIM, s.z + HALF, E_DIM, s.up1, out_ld, s, stream);
  // coupling 2: cond = up1, tgt = original lower = z[:, :32]       -> lo1
  run_coupling(f2, s.up1, HALF, s.z, E_DIM, s.lo1, out_ld, s, stream);

  // concat [lower, upper]
  k_pack<<<dim3(MROWS * E_DIM / 256), dim3(256), 0, stream>>>(s.lo1, s.up1, out_z, MROWS);
}
